// DynamicQueryAttention_36842229465483
// MI455X (gfx1250) — matrix operand order repaired and checked by reading, not compiled, not run on hardware
//
#include <hip/hip_runtime.h>
#include <hip/hip_bf16.h>
#include <stdint.h>

#define D_MODEL 1024
#define N_HEADS 16
#define D_HEAD  64
#define BATCH   2
#define T_SEQ   2048
#define ROWS    (BATCH * T_SEQ)   // 4096

typedef __attribute__((ext_vector_type(16))) __bf16 v16bf;
typedef __attribute__((ext_vector_type(8)))  float  v8f;

union FragBF { v16bf v; uint4 q[2]; };

__device__ __forceinline__ unsigned short f32_to_bf16_bits(float f) {
  union { float f; uint32_t u; } c; c.f = f;
  uint32_t u = c.u;
  u += 0x7fffu + ((u >> 16) & 1u);           // round-to-nearest-even
  return (unsigned short)(u >> 16);
}
__device__ __forceinline__ float bf16_bits_to_f32(unsigned short h) {
  union { uint32_t u; float f; } c; c.u = ((uint32_t)h) << 16;
  return c.f;
}

__device__ __forceinline__ void load_fragA(FragBF& f, const unsigned short* arow,
                                           int k0, int lo) {
  f.q[0] = *(const uint4*)(arow + k0 + 2 * lo);
  f.q[1] = *(const uint4*)(arow + k0 + 2 * lo + 8);
}
__device__ __forceinline__ void load_fragB(FragBF& f, const unsigned short* wp) {
  f.q[0] = *(const uint4*)(wp);
  f.q[1] = *(const uint4*)(wp + 8);
}

// ---------------------------------------------------------------- convert (x4)
__global__ void cvt_f32_to_bf16_x4(const float* __restrict__ in,
                                   unsigned short* __restrict__ out, int n4) {
  int i = blockIdx.x * blockDim.x + threadIdx.x;
  if (i >= n4) return;
  float4 f = ((const float4*)in)[i];
  union { unsigned short s[4]; uint2 u; } o;
  o.s[0] = f32_to_bf16_bits(f.x);
  o.s[1] = f32_to_bf16_bits(f.y);
  o.s[2] = f32_to_bf16_bits(f.z);
  o.s[3] = f32_to_bf16_bits(f.w);
  ((uint2*)out)[i] = o.u;
}

// ---------------------------------------------------------------- gate MLP
// gate = sigmoid(silu(x @ Wg1^T) @ Wg2^T + bg2)   -> (ROWS, N_HEADS) fp32
__global__ __launch_bounds__(256) void gate_kernel(
    const float* __restrict__ x, const float* __restrict__ Wg1,
    const float* __restrict__ Wg2, const float* __restrict__ bg2,
    float* __restrict__ gate) {
  __shared__ float hbuf[16][16];
  int tid = threadIdx.x;
  int rl = tid >> 4, g = tid & 15;
  int row = blockIdx.x * 16 + rl;
  const float* xr = x + (size_t)row * D_MODEL;
  const float* wr = Wg1 + (size_t)g * D_MODEL;
  float acc = 0.f;
  for (int k = 0; k < D_MODEL; k += 4) {
    acc += xr[k] * wr[k] + xr[k + 1] * wr[k + 1]
         + xr[k + 2] * wr[k + 2] + xr[k + 3] * wr[k + 3];
  }
  hbuf[rl][g] = acc / (1.f + __expf(-acc));   // silu
  __syncthreads();
  float s2 = bg2[g];
  const float* w2 = Wg2 + g * 16;
  #pragma unroll
  for (int j = 0; j < 16; ++j) s2 += w2[j] * hbuf[rl][j];
  gate[row * 16 + g] = 1.f / (1.f + __expf(-s2));
}

// ---------------------------------------------------------------- WMMA GEMM
// C[m][n] = sum_k A[m][k] * W[n][k]   (A: ROWS x 1024 bf16, W: 1024 x 1024 bf16)
// mode 0: store fp32 row-major; mode 1: store bf16 row-major;
// mode 2: store bf16 V-transposed as (BATCH, 1024, T_SEQ)
// Double-buffered K-loop so next fragments load while current WMMAs execute.
__global__ __launch_bounds__(256) void gemm_bf16_wmma(
    const unsigned short* __restrict__ A,
    const unsigned short* __restrict__ W,
    float* __restrict__ Cf, unsigned short* __restrict__ Cb, int mode) {
  int lane = threadIdx.x & 31;
  int wv   = threadIdx.x >> 5;
  int gw   = blockIdx.x * 8 + wv;      // 0..4095
  int tn4  = gw & 15;                  // 64-column group
  int tm   = gw >> 4;                  // 16-row tile, 0..255
  int lanelo = lane & 15;
  int half   = lane >> 4;
  int lo     = half * 8;               // A-frag K chunk base is 2*lo
  int kb16   = half * 16;              // B-frag K chunk base

  const unsigned short* arow = A + (size_t)(tm * 16 + lanelo) * D_MODEL;
  const unsigned short* wbase[4];
  #pragma unroll
  for (int t = 0; t < 4; ++t)
    wbase[t] = W + (size_t)(tn4 * 64 + t * 16 + lanelo) * D_MODEL + kb16;

  v8f acc[4] = {};
  FragBF a0, a1, b0[4], b1[4];
  load_fragA(a0, arow, 0, lo);
  #pragma unroll
  for (int t = 0; t < 4; ++t) load_fragB(b0[t], wbase[t]);

  for (int k0 = 0; k0 < D_MODEL; k0 += 64) {
    // stage 1: issue loads for k0+32, compute with k0
    load_fragA(a1, arow, k0 + 32, lo);
    #pragma unroll
    for (int t = 0; t < 4; ++t) load_fragB(b1[t], wbase[t] + k0 + 32);
    if (k0 + 64 < D_MODEL) __builtin_prefetch(arow + k0 + 64 + lo, 0, 1);
    #pragma unroll
    for (int t = 0; t < 4; ++t)
      acc[t] = __builtin_amdgcn_wmma_f32_16x16x32_bf16(
          false, a0.v, false, b0[t].v, (short)0, acc[t], false, false);
    // stage 2: issue loads for k0+64, compute with k0+32
    if (k0 + 64 < D_MODEL) {
      load_fragA(a0, arow, k0 + 64, lo);
      #pragma unroll
      for (int t = 0; t < 4; ++t) load_fragB(b0[t], wbase[t] + k0 + 64);
    }
    #pragma unroll
    for (int t = 0; t < 4; ++t)
      acc[t] = __builtin_amdgcn_wmma_f32_16x16x32_bf16(
          false, a1.v, false, b1[t].v, (short)0, acc[t], false, false);
  }

  #pragma unroll
  for (int t = 0; t < 4; ++t) {
    int col = tn4 * 64 + t * 16 + lanelo;
    #pragma unroll
    for (int r = 0; r < 8; ++r) {
      int m = tm * 16 + r + 8 * half;
      float val = acc[t][r];
      if (mode == 0) {
        Cf[(size_t)m * D_MODEL + col] = val;
      } else if (mode == 1) {
        Cb[(size_t)m * D_MODEL + col] = f32_to_bf16_bits(val);
      } else {
        int b = m >> 11, s = m & (T_SEQ - 1);
        Cb[((size_t)(b * 1024 + col)) * T_SEQ + s] = f32_to_bf16_bits(val);
      }
    }
  }
}

// ---------------------------------------------------------------- Q combine
__global__ void combine_q(const unsigned short* __restrict__ Qb,
                          const unsigned short* __restrict__ Qd,
                          const float* __restrict__ gate,
                          unsigned short* __restrict__ Q) {
  int i = blockIdx.x * blockDim.x + threadIdx.x;
  if (i >= ROWS * D_MODEL) return;
  int row = i >> 10, col = i & 1023, h = col >> 6;
  float q = bf16_bits_to_f32(Qb[i]) + gate[row * 16 + h] * bf16_bits_to_f32(Qd[i]);
  Q[i] = f32_to_bf16_bits(q);
}

// ---------------------------------------------------------------- flash attention
// one wave = one (b, h, 16-query tile); K layout (ROWS,1024) bf16; Vt (BATCH,1024,T)
__global__ __launch_bounds__(256) void attn_kernel(
    const unsigned short* __restrict__ Q,
    const unsigned short* __restrict__ K,
    const unsigned short* __restrict__ Vt,
    unsigned short* __restrict__ O) {
  __shared__ __attribute__((aligned(16))) unsigned short ldsP[8][16 * 32];
  int lane = threadIdx.x & 31;
  int wv   = threadIdx.x >> 5;
  int gw   = blockIdx.x * 8 + wv;      // 0..4095
  int qt   = gw & 127;
  int h    = (gw >> 7) & 15;
  int b    = gw >> 11;
  int q0   = qt * 16;
  int lanelo = lane & 15;
  int half   = lane >> 4;
  int lo     = half * 8;

  // Q A-fragments covering d = 0..63 (two K=32 steps)
  const unsigned short* qrow =
      Q + ((size_t)(b * T_SEQ + q0 + lanelo)) * D_MODEL + h * D_HEAD;
  FragBF qa0, qa1;
  qa0.q[0] = *(const uint4*)(qrow + 2 * lo);
  qa0.q[1] = *(const uint4*)(qrow + 2 * lo + 8);
  qa1.q[0] = *(const uint4*)(qrow + 32 + 2 * lo);
  qa1.q[1] = *(const uint4*)(qrow + 40 + 2 * lo);

  v8f acc[4] = {};
  float rowmax[8], rowsum[8];
  #pragma unroll
  for (int r = 0; r < 8; ++r) { rowmax[r] = -1e30f; rowsum[r] = 0.f; }

  const float inv_scale = 0.125f;            // 1/sqrt(64)
  int npair = (q0 + 16 + 31) >> 5;           // 32-key pairs to process

  const unsigned short* kb_base = K + ((size_t)(b * T_SEQ)) * D_MODEL + h * D_HEAD;
  const unsigned short* vt_base = Vt + ((size_t)(b * 1024 + h * D_HEAD)) * T_SEQ;

  for (int j = 0; j < npair; ++j) {
    int key0 = j * 32;

    // K B-fragments for both 16-key sub-blocks (d-steps 0 and 1)
    FragBF kf[2][2];
    #pragma unroll
    for (int sub = 0; sub < 2; ++sub) {
      const unsigned short* kp =
          kb_base + (size_t)(key0 + sub * 16 + lanelo) * D_MODEL + half * 16;
      load_fragB(kf[sub][0], kp);
      load_fragB(kf[sub][1], kp + 32);
    }
    // V B-fragments issued early: they overlap the softmax VALU work below
    FragBF vf[4];
    #pragma unroll
    for (int t = 0; t < 4; ++t) {
      const unsigned short* vp =
          vt_base + (size_t)(t * 16 + lanelo) * T_SEQ + key0 + half * 16;
      load_fragB(vf[t], vp);
    }
    if (j + 1 < npair)
      __builtin_prefetch(kb_base + (size_t)(key0 + 32 + lanelo) * D_MODEL, 0, 1);

    v8f s[2];
    #pragma unroll
    for (int sub = 0; sub < 2; ++sub) {
      v8f sv = {};
      sv = __builtin_amdgcn_wmma_f32_16x16x32_bf16(
          false, qa0.v, false, kf[sub][0].v, (short)0, sv, false, false);
      sv = __builtin_amdgcn_wmma_f32_16x16x32_bf16(
          false, qa1.v, false, kf[sub][1].v, (short)0, sv, false, false);
      s[sub] = sv;
    }

    // scale + causal mask + joint online softmax over the 32 keys
    float factor[8], pa[8], pb[8];
    #pragma unroll
    for (int r = 0; r < 8; ++r) {
      int qidx = q0 + r + 8 * half;
      float va = ((key0 + lanelo)      <= qidx) ? s[0][r] * inv_scale : -1e30f;
      float vb = ((key0 + 16 + lanelo) <= qidx) ? s[1][r] * inv_scale : -1e30f;
      float mx = fmaxf(va, vb);
      #pragma unroll
      for (int m = 1; m < 16; m <<= 1) mx = fmaxf(mx, __shfl_xor(mx, m, 16));
      float nm = fmaxf(rowmax[r], mx);
      float f  = __expf(rowmax[r] - nm);
      rowmax[r] = nm;
      factor[r] = f;
      float ea = __expf(va - nm);
      float eb = __expf(vb - nm);
      pa[r] = ea; pb[r] = eb;
      float ps = ea + eb;
      #pragma unroll
      for (int m = 1; m < 16; m <<= 1) ps += __shfl_xor(ps, m, 16);
      rowsum[r] = rowsum[r] * f + ps;
    }

    // stage P (bf16) through per-wave LDS to re-layout C-tile -> A-fragment
    unsigned short* lp = &ldsP[wv][0];
    #pragma unroll
    for (int r = 0; r < 8; ++r) {
      int m = r + 8 * half;
      lp[m * 32 + lanelo]      = f32_to_bf16_bits(pa[r]);
      lp[m * 32 + 16 + lanelo] = f32_to_bf16_bits(pb[r]);
      #pragma unroll
      for (int t = 0; t < 4; ++t) acc[t][r] *= factor[r];
    }
    asm volatile("s_wait_dscnt 0x0" ::: "memory");

    FragBF pf;
    const unsigned short* lrow = &ldsP[wv][lanelo * 32];
    pf.q[0] = *(const uint4*)(lrow + 2 * lo);
    pf.q[1] = *(const uint4*)(lrow + 2 * lo + 8);

    // O += P x V  (V fragments already in registers)
    #pragma unroll
    for (int t = 0; t < 4; ++t)
      acc[t] = __builtin_amdgcn_wmma_f32_16x16x32_bf16(
          false, pf.v, false, vf[t].v, (short)0, acc[t], false, false);
  }

  // epilogue: normalize and store bf16 attention output (ROWS,1024)
  #pragma unroll
  for (int t = 0; t < 4; ++t) {
    #pragma unroll
    for (int r = 0; r < 8; ++r) {
      int m = q0 + r + 8 * half;
      float o = acc[t][r] / rowsum[r];
      O[((size_t)(b * T_SEQ + m)) * D_MODEL + h * D_HEAD + t * 16 + lanelo] =
          f32_to_bf16_bits(o);
    }
  }
}

// ---------------------------------------------------------------- launch
extern "C" void kernel_launch(void* const* d_in, const int* in_sizes, int n_in,
                              void* d_out, int out_size, void* d_ws, size_t ws_size,
                              hipStream_t stream) {
  const float* x   = (const float*)d_in[0];
  // d_in[1] = mask (bool) — causality handled analytically, unused
  const float* Wk  = (const float*)d_in[2];
  const float* Wv  = (const float*)d_in[3];
  const float* Wo  = (const float*)d_in[4];
  const float* Wqb = (const float*)d_in[5];
  const float* Wqd = (const float*)d_in[6];
  const float* Wg1 = (const float*)d_in[7];
  const float* Wg2 = (const float*)d_in[8];
  const float* bg2 = (const float*)d_in[9];
  float* out = (float*)d_out;

  char* ws = (char*)d_ws;
  const size_t SZ_X  = (size_t)ROWS * D_MODEL * 2;       // 8 MB
  const size_t SZ_W  = (size_t)D_MODEL * D_MODEL * 2;    // 2 MB
  unsigned short* Xbf   = (unsigned short*)(ws);
  unsigned short* WkB   = (unsigned short*)(ws + SZ_X);
  unsigned short* WvB   = (unsigned short*)(ws + SZ_X + SZ_W);
  unsigned short* WqbB  = (unsigned short*)(ws + SZ_X + 2 * SZ_W);
  unsigned short* WqdB  = (unsigned short*)(ws + SZ_X + 3 * SZ_W);
  unsigned short* WoB   = (unsigned short*)(ws + SZ_X + 4 * SZ_W);
  unsigned short* Kbf   = (unsigned short*)(ws + SZ_X + 5 * SZ_W);
  unsigned short* Vtbf  = (unsigned short*)(ws + 2 * SZ_X + 5 * SZ_W);
  unsigned short* QbB   = (unsigned short*)(ws + 3 * SZ_X + 5 * SZ_W);
  unsigned short* QdB   = (unsigned short*)(ws + 4 * SZ_X + 5 * SZ_W);
  unsigned short* Qbf   = (unsigned short*)(ws + 5 * SZ_X + 5 * SZ_W);
  float*          gateB = (float*)(ws + 6 * SZ_X + 5 * SZ_W);
  unsigned short* AttnB = QbB;   // reuse: Qb dead after combine_q

  const int NX4 = ROWS * D_MODEL / 4;       // 1048576
  const int NW4 = D_MODEL * D_MODEL / 4;    // 262144

  cvt_f32_to_bf16_x4<<<(NX4 + 255) / 256, 256, 0, stream>>>(x,   Xbf,  NX4);
  cvt_f32_to_bf16_x4<<<(NW4 + 255) / 256, 256, 0, stream>>>(Wk,  WkB,  NW4);
  cvt_f32_to_bf16_x4<<<(NW4 + 255) / 256, 256, 0, stream>>>(Wv,  WvB,  NW4);
  cvt_f32_to_bf16_x4<<<(NW4 + 255) / 256, 256, 0, stream>>>(Wqb, WqbB, NW4);
  cvt_f32_to_bf16_x4<<<(NW4 + 255) / 256, 256, 0, stream>>>(Wqd, WqdB, NW4);
  cvt_f32_to_bf16_x4<<<(NW4 + 255) / 256, 256, 0, stream>>>(Wo,  WoB,  NW4);

  gate_kernel<<<ROWS / 16, 256, 0, stream>>>(x, Wg1, Wg2, bg2, gateB);

  // projections (512 blocks x 8 waves = 4096 16x64 tiles)
  gemm_bf16_wmma<<<512, 256, 0, stream>>>(Xbf, WkB,  nullptr, Kbf,  1);
  gemm_bf16_wmma<<<512, 256, 0, stream>>>(Xbf, WvB,  nullptr, Vtbf, 2);
  gemm_bf16_wmma<<<512, 256, 0, stream>>>(Xbf, WqbB, nullptr, QbB,  1);
  gemm_bf16_wmma<<<512, 256, 0, stream>>>(Xbf, WqdB, nullptr, QdB,  1);

  combine_q<<<(ROWS * D_MODEL + 255) / 256, 256, 0, stream>>>(QbB, QdB, gateB, Qbf);

  attn_kernel<<<512, 256, 0, stream>>>(Qbf, Kbf, Vtbf, AttnB);

  gemm_bf16_wmma<<<512, 256, 0, stream>>>(AttnB, WoB, out, nullptr, 0);
}
